// CGMPBlock_71932112273871
// MI455X (gfx1250) — matrix operand order repaired and checked by reading, not compiled, not run on hardware
//
#include <hip/hip_runtime.h>

#define NN 4096
#define EE 65536

typedef __attribute__((ext_vector_type(16))) _Float16 v16h;
typedef __attribute__((ext_vector_type(2)))  _Float16 h2;
typedef __attribute__((ext_vector_type(8)))  float    v8f;

#define RS2C 0.70710678118654752f
#define RS3C 0.57735026918962576f

// ---------------- workspace layout (bytes) ----------------
// y0 : [2][N][32]    f32   @ 0          (1,048,576)
// y1 : [2][N][3][32] f32   @ 1,048,576  (3,145,728)
// Wt0: [3][16][3584] f16   @ 4,194,304  (344,064)  planes: Wr, Wi, -Wi (scales folded)
// Wt1: [3][16][5376] f16   @ 4,538,368  (516,096)

// B-side LDS strides in f16 elements (multiples of 8 -> 16B-aligned pair runs)
#define H_Y0 40   // logical 32
#define H_Y1 104  // logical 96 (3*32)
#define H_X0 24   // logical 16
#define H_X1 56   // logical 48 (3*16)
// A-side duplicated-pair LDS strides in h2 units (1 dword per channel)
#define D_Y0 34
#define D_Y1 100
#define D_X0 18
#define D_X1 50

union F16x16 { v16h v; h2 p[8]; };

__device__ __forceinline__ v8f wmma16(v16h a, v16h b, v8f c) {
  return __builtin_amdgcn_wmma_f32_16x16x32_f16(false, a, false, b, (short)0, c,
                                                false, false);
}

__device__ __forceinline__ v16h load_bfrag(const _Float16* __restrict__ p) {
  union { uint4 q[2]; v16h v; } u;
  u.q[0] = *reinterpret_cast<const uint4*>(p);
  u.q[1] = *reinterpret_cast<const uint4*>(p + 16);
  return u.v;
}

// --------- packed-f16 fragment builders: 4 pairs (8 elems) at pair-ofs p0 ----
// z = A * B[db..]
__device__ __forceinline__ void put8_1(F16x16& fr, F16x16& fi, int p0,
    h2 Ar, h2 Ai,
    const _Float16* __restrict__ Br, const _Float16* __restrict__ Bi, int db) {
  const h2* br = (const h2*)(Br + db);
  const h2* bi = (const h2*)(Bi + db);
#pragma unroll
  for (int p = 0; p < 4; ++p) {
    h2 b_r = br[p], b_i = bi[p];
    fr.p[p0 + p] = Ar * b_r - Ai * b_i;
    fi.p[p0 + p] = Ar * b_i + Ai * b_r;
  }
}

// z = A1*B1[..] - A2*B2[..]        (antisymmetric 2-term; scale folded in W)
__device__ __forceinline__ void put8_2(F16x16& fr, F16x16& fi, int p0,
    h2 a1r, h2 a1i, const _Float16* __restrict__ B1r, const _Float16* __restrict__ B1i,
    h2 a2r, h2 a2i, const _Float16* __restrict__ B2r, const _Float16* __restrict__ B2i,
    int db) {
  const h2* b1r = (const h2*)(B1r + db); const h2* b1i = (const h2*)(B1i + db);
  const h2* b2r = (const h2*)(B2r + db); const h2* b2i = (const h2*)(B2i + db);
#pragma unroll
  for (int p = 0; p < 4; ++p) {
    h2 u_r = b1r[p], u_i = b1i[p];
    h2 v_r = b2r[p], v_i = b2i[p];
    fr.p[p0 + p] = (a1r * u_r - a1i * u_i) - (a2r * v_r - a2i * v_i);
    fi.p[p0 + p] = (a1r * u_i + a1i * u_r) - (a2r * v_i + a2i * v_r);
  }
}

// z = A1*B1[..] - A2*B2[..] + A3*B3[..]   (3-term; scale folded in W)
__device__ __forceinline__ void put8_3(F16x16& fr, F16x16& fi, int p0,
    h2 a1r, h2 a1i, const _Float16* __restrict__ B1r, const _Float16* __restrict__ B1i,
    h2 a2r, h2 a2i, const _Float16* __restrict__ B2r, const _Float16* __restrict__ B2i,
    h2 a3r, h2 a3i, const _Float16* __restrict__ B3r, const _Float16* __restrict__ B3i,
    int db) {
  const h2* b1r = (const h2*)(B1r + db); const h2* b1i = (const h2*)(B1i + db);
  const h2* b2r = (const h2*)(B2r + db); const h2* b2i = (const h2*)(B2i + db);
  const h2* b3r = (const h2*)(B3r + db); const h2* b3i = (const h2*)(B3i + db);
#pragma unroll
  for (int p = 0; p < 4; ++p) {
    h2 u_r = b1r[p], u_i = b1i[p];
    h2 v_r = b2r[p], v_i = b2i[p];
    h2 w_r = b3r[p], w_i = b3i[p];
    fr.p[p0 + p] = (a1r * u_r - a1i * u_i) - (a2r * v_r - a2i * v_i) +
                   (a3r * w_r - a3i * w_i);
    fi.p[p0 + p] = (a1r * u_i + a1i * u_r) - (a2r * v_i + a2i * v_r) +
                   (a3r * w_i + a3i * w_r);
  }
}

// complex accumulate: 4 WMMAs, imaginary-weight negation pre-folded (bn = -bi)
__device__ __forceinline__ void accum_c(v8f& accR, v8f& accI,
                                        v16h ar, v16h ai,
                                        v16h br, v16h bi, v16h bn) {
  accR = wmma16(ar, br, accR);
  accR = wmma16(ai, bn, accR);
  accI = wmma16(ar, bi, accI);
  accI = wmma16(ai, br, accI);
}

// ---------------------------------------------------------------------------
// Edge message passing: y[l][ri, dst, m, k*16+c] += x[l][ri, src, m, c]*ev[e,k]
// ---------------------------------------------------------------------------
__global__ void k_edge_mp(const float* __restrict__ x0, const float* __restrict__ x1,
                          const float* __restrict__ ev, const int* __restrict__ eidx,
                          float* __restrict__ y0, float* __restrict__ y1) {
  int i = blockIdx.x * blockDim.x + threadIdx.x;   // over EE*16
  int e = i >> 4, c = i & 15;
  int s = eidx[e], d = eidx[EE + e];
  float v0 = ev[e * 2 + 0], v1 = ev[e * 2 + 1];
#pragma unroll
  for (int ri = 0; ri < 2; ++ri) {
    float a = x0[(ri * NN + s) * 16 + c];
    atomicAdd(&y0[(ri * NN + d) * 32 + c], a * v0);
    atomicAdd(&y0[(ri * NN + d) * 32 + 16 + c], a * v1);
#pragma unroll
    for (int m = 0; m < 3; ++m) {
      float b = x1[((ri * NN + s) * 3 + m) * 16 + c];
      atomicAdd(&y1[((ri * NN + d) * 3 + m) * 32 + c], b * v0);
      atomicAdd(&y1[((ri * NN + d) * 3 + m) * 32 + 16 + c], b * v1);
    }
  }
}

// ---------------------------------------------------------------------------
// Weight repack (f16, [plane][f][K], planes Wr / Wi / -Wi; CG scales folded).
// ---------------------------------------------------------------------------
__global__ void k_prep_w0(const float* __restrict__ wmm0r, const float* __restrict__ wmm0i,
                          const float* __restrict__ wmi0r, const float* __restrict__ wmi0i,
                          const float* __restrict__ wii0r, const float* __restrict__ wii0i,
                          _Float16* __restrict__ Wt0) {
  int i = blockIdx.x * blockDim.x + threadIdx.x;
  if (i >= 3 * 16 * 3584) return;
  int kk = i % 3584, rf = i / 3584, f = rf & 15, pl = rf / 16;
  int im = (pl >= 1);
  const float* src; int row; float sc = 1.f;
  if (kk < 2048)      { src = im ? wmm0i : wmm0r; row = kk;        if (kk >= 1024) sc = RS3C; }
  else if (kk < 3072) { src = im ? wmi0i : wmi0r; row = kk - 2048; if (kk >= 2560) sc = RS3C; }
  else                { src = im ? wii0i : wii0r; row = kk - 3072; if (kk >= 3328) sc = RS3C; }
  float v = sc * src[row * 16 + f];
  if (pl == 2) v = -v;
  Wt0[(pl * 16 + f) * 3584 + kk] = (_Float16)v;
}

__global__ void k_prep_w1(const float* __restrict__ wmm1r, const float* __restrict__ wmm1i,
                          const float* __restrict__ wmi1r, const float* __restrict__ wmi1i,
                          const float* __restrict__ wii1r, const float* __restrict__ wii1i,
                          _Float16* __restrict__ Wt1) {
  int i = blockIdx.x * blockDim.x + threadIdx.x;
  if (i >= 3 * 16 * 5376) return;
  int kk = i % 5376, rf = i / 5376, f = rf & 15, pl = rf / 16;
  int im = (pl >= 1);
  const float* src; int row; float sc = 1.f;
  if (kk < 3072)      { src = im ? wmm1i : wmm1r; row = kk;        if (kk >= 2048) sc = RS2C; }
  else if (kk < 4608) { src = im ? wmi1i : wmi1r; row = kk - 3072; if (kk >= 4096) sc = RS2C; }
  else                { src = im ? wii1i : wii1r; row = kk - 4608; if (kk >= 5120) sc = RS2C; }
  float v = sc * src[row * 16 + f];
  if (pl == 2) v = -v;
  Wt1[(pl * 16 + f) * 5376 + kk] = (_Float16)v;
}

// ---------------------------------------------------------------------------
// Stage features into LDS twice: f16 B-side (consecutive-d runs) and
// pre-duplicated h2 A-side (ds_load_b32 == splatted operand).
// ---------------------------------------------------------------------------
__device__ __forceinline__ void stage_features(
    int tid, int nb,
    const float* __restrict__ x0, const float* __restrict__ x1,
    const float* __restrict__ y0g, const float* __restrict__ y1g,
    _Float16* __restrict__ sy0, _Float16* __restrict__ sy1,
    _Float16* __restrict__ sx0, _Float16* __restrict__ sx1,
    h2* __restrict__ dy0, h2* __restrict__ dy1,
    h2* __restrict__ dx0, h2* __restrict__ dx1) {
  for (int i = tid; i < 2 * 16 * 32; i += 128) {        // y0
    int ri = i >> 9, r = i & 511, tt = r >> 5, c = r & 31;
    _Float16 h = (_Float16)y0g[(ri * NN + nb + tt) * 32 + c];
    sy0[(ri * 16 + tt) * H_Y0 + c] = h;
    h2 d; d.x = h; d.y = h;
    dy0[(ri * 16 + tt) * D_Y0 + c] = d;
  }
  for (int i = tid; i < 2 * 16 * 96; i += 128) {        // y1
    int ri = i / (16 * 96), r = i % (16 * 96), tt = r / 96, mc = r % 96;
    _Float16 h = (_Float16)y1g[(ri * NN + nb + tt) * 96 + mc];
    sy1[(ri * 16 + tt) * H_Y1 + mc] = h;
    h2 d; d.x = h; d.y = h;
    dy1[(ri * 16 + tt) * D_Y1 + mc] = d;
  }
  for (int i = tid; i < 2 * 16 * 16; i += 128) {        // x0
    int ri = i >> 8, r = i & 255, tt = r >> 4, c = r & 15;
    _Float16 h = (_Float16)x0[(ri * NN + nb + tt) * 16 + c];
    sx0[(ri * 16 + tt) * H_X0 + c] = h;
    h2 d; d.x = h; d.y = h;
    dx0[(ri * 16 + tt) * D_X0 + c] = d;
  }
  for (int i = tid; i < 2 * 16 * 48; i += 128) {        // x1
    int ri = i / (16 * 48), r = i % (16 * 48), tt = r / 48, mc = r % 48;
    _Float16 h = (_Float16)x1[(ri * NN + nb + tt) * 48 + mc];
    sx1[(ri * 16 + tt) * H_X1 + mc] = h;
    h2 d; d.x = h; d.y = h;
    dx1[(ri * 16 + tt) * D_X1 + mc] = d;
  }
}

// ---------------------------------------------------------------------------
// Fused CG-expansion + complex GEMM, l = 0  (D = 3584).
// ---------------------------------------------------------------------------
__global__ __launch_bounds__(128) void k_cg_l0(
    const float* __restrict__ x0, const float* __restrict__ x1,
    const float* __restrict__ y0g, const float* __restrict__ y1g,
    const _Float16* __restrict__ Wt0, float* __restrict__ out0) {
  __shared__ _Float16 sy0[2 * 16 * H_Y0];
  __shared__ _Float16 sy1[2 * 16 * H_Y1];
  __shared__ _Float16 sx0[2 * 16 * H_X0];
  __shared__ _Float16 sx1[2 * 16 * H_X1];
  __shared__ h2 dy0[2 * 16 * D_Y0];
  __shared__ h2 dy1[2 * 16 * D_Y1];
  __shared__ h2 dx0[2 * 16 * D_X0];
  __shared__ h2 dx1[2 * 16 * D_X1];
  __shared__ float sred[4 * 2 * 8 * 32];

  const int tid = threadIdx.x;
  const int wu = __builtin_amdgcn_readfirstlane(tid >> 5);  // wave id (scalar)
  const int ln = tid & 31;
  const int t = ln & 15, half = ln >> 4;
  const int nb = blockIdx.x * 16;

  stage_features(tid, nb, x0, x1, y0g, y1g, sy0, sy1, sx0, sx1, dy0, dy1, dx0, dx1);
  __syncthreads();

  const _Float16* y0r = sy0 + t * H_Y0;  const _Float16* y0i = sy0 + (16 + t) * H_Y0;
  const _Float16* y1r = sy1 + t * H_Y1;  const _Float16* y1i = sy1 + (16 + t) * H_Y1;
  const _Float16* x0r = sx0 + t * H_X0;  const _Float16* x0i = sx0 + (16 + t) * H_X0;
  const _Float16* x1r = sx1 + t * H_X1;  const _Float16* x1i = sx1 + (16 + t) * H_X1;
  const h2* Y0r = dy0 + t * D_Y0;  const h2* Y0i = dy0 + (16 + t) * D_Y0;
  const h2* Y1r = dy1 + t * D_Y1;  const h2* Y1i = dy1 + (16 + t) * D_Y1;
  const h2* X0r = dx0 + t * D_X0;  const h2* X0i = dx0 + (16 + t) * D_X0;
  const h2* X1r = dx1 + t * D_X1;  const h2* X1i = dx1 + (16 + t) * D_X1;

  v8f accR = {}, accI = {};
  const _Float16* wrb = Wt0 + (0 * 16 + t) * 3584;
  const _Float16* wib = Wt0 + (1 * 16 + t) * 3584;
  const _Float16* wnb = Wt0 + (2 * 16 + t) * 3584;
  const int kselB = half * 8;
  const int db = half * 8;

  for (int kb = wu * 32; kb < 3584; kb += 128) {   // kb is scalar
    v16h br = load_bfrag(wrb + kb + kselB);
    v16h bi = load_bfrag(wib + kb + kselB);
    v16h bn = load_bfrag(wnb + kb + kselB);
    F16x16 far_, fai_;
    if (kb < 1024) {                        // yy(0,0): wide, A=y0[c], B=y0
      int c = kb >> 5;
      h2 Ar = Y0r[c], Ai = Y0i[c];
      put8_1(far_, fai_, 0, Ar, Ai, y0r, y0i, db);
      put8_1(far_, fai_, 4, Ar, Ai, y0r, y0i, 16 + db);
    } else if (kb < 2048) {                 // yy(1,1): wide 3-term (scale in W)
      int c = (kb - 1024) >> 5;
      h2 a1r = Y1r[c],      a1i = Y1i[c];
      h2 a2r = Y1r[32 + c], a2i = Y1i[32 + c];
      h2 a3r = Y1r[64 + c], a3i = Y1i[64 + c];
      put8_3(far_, fai_, 0, a1r, a1i, y1r + 64, y1i + 64,
                            a2r, a2i, y1r + 32, y1i + 32,
                            a3r, a3i, y1r, y1i, db);
      put8_3(far_, fai_, 4, a1r, a1i, y1r + 64, y1i + 64,
                            a2r, a2i, y1r + 32, y1i + 32,
                            a3r, a3i, y1r, y1i, 16 + db);
    } else if (kb < 2560) {                 // yx(0,0): narrow, A=y0[c], B=x0
      int c = (kb - 2048) >> 4;
      put8_1(far_, fai_, 0, Y0r[c],     Y0i[c],     x0r, x0i, db);
      put8_1(far_, fai_, 4, Y0r[c + 1], Y0i[c + 1], x0r, x0i, db);
    } else if (kb < 3072) {                 // yx(1,1): narrow 3-term
      int c = (kb - 2560) >> 4;
#pragma unroll
      for (int h = 0; h < 2; ++h) {
        int cc = c + h;
        put8_3(far_, fai_, h * 4,
               Y1r[cc],      Y1i[cc],      x1r + 32, x1i + 32,
               Y1r[32 + cc], Y1i[32 + cc], x1r + 16, x1i + 16,
               Y1r[64 + cc], Y1i[64 + cc], x1r, x1i, db);
      }
    } else if (kb < 3328) {                 // xx(0,0): narrow
      int c = (kb - 3072) >> 4;
      put8_1(far_, fai_, 0, X0r[c],     X0i[c],     x0r, x0i, db);
      put8_1(far_, fai_, 4, X0r[c + 1], X0i[c + 1], x0r, x0i, db);
    } else {                                // xx(1,1): narrow 3-term
      int c = (kb - 3328) >> 4;
#pragma unroll
      for (int h = 0; h < 2; ++h) {
        int cc = c + h;
        put8_3(far_, fai_, h * 4,
               X1r[cc],      X1i[cc],      x1r + 32, x1i + 32,
               X1r[16 + cc], X1i[16 + cc], x1r + 16, x1i + 16,
               X1r[32 + cc], X1i[32 + cc], x1r, x1i, db);
      }
    }
    accum_c(accR, accI, far_.v, fai_.v, br, bi, bn);
  }

  v8f acc[2] = { accR, accI };
#pragma unroll
  for (int a = 0; a < 2; ++a)
#pragma unroll
    for (int v = 0; v < 8; ++v)
      sred[((wu * 2 + a) * 8 + v) * 32 + ln] = acc[a][v];
  __syncthreads();
  if (wu == 0) {
#pragma unroll
    for (int a = 0; a < 2; ++a)
#pragma unroll
      for (int v = 0; v < 8; ++v) {
        float s = 0.f;
#pragma unroll
        for (int ww = 0; ww < 4; ++ww) s += sred[((ww * 2 + a) * 8 + v) * 32 + ln];
        int node = nb + v + 8 * half;
        out0[(a * NN + node) * 16 + t] = s;
      }
  }
}

// ---------------------------------------------------------------------------
// Fused CG-expansion + complex GEMM, l = 1 (D = 5376).  B fragments reused
// across the three q rows.  (1,1,1) CG index pairs (i1,i2): q=0:(1,0),
// q=1:(2,0), q=2:(2,1);  z = A[i1]*B[i2] - A[i2]*B[i1]  (1/sqrt2 in W).
// ---------------------------------------------------------------------------
__global__ __launch_bounds__(128) void k_cg_l1(
    const float* __restrict__ x0, const float* __restrict__ x1,
    const float* __restrict__ y0g, const float* __restrict__ y1g,
    const _Float16* __restrict__ Wt1, float* __restrict__ out1) {
  __shared__ _Float16 sy0[2 * 16 * H_Y0];
  __shared__ _Float16 sy1[2 * 16 * H_Y1];
  __shared__ _Float16 sx0[2 * 16 * H_X0];
  __shared__ _Float16 sx1[2 * 16 * H_X1];
  __shared__ h2 dy0[2 * 16 * D_Y0];
  __shared__ h2 dy1[2 * 16 * D_Y1];
  __shared__ h2 dx0[2 * 16 * D_X0];
  __shared__ h2 dx1[2 * 16 * D_X1];
  __shared__ float sred[4 * 6 * 8 * 32];

  const int tid = threadIdx.x;
  const int wu = __builtin_amdgcn_readfirstlane(tid >> 5);
  const int ln = tid & 31;
  const int t = ln & 15, half = ln >> 4;
  const int nb = blockIdx.x * 16;

  stage_features(tid, nb, x0, x1, y0g, y1g, sy0, sy1, sx0, sx1, dy0, dy1, dx0, dx1);
  __syncthreads();

  const _Float16* y0r = sy0 + t * H_Y0;  const _Float16* y0i = sy0 + (16 + t) * H_Y0;
  const _Float16* y1r = sy1 + t * H_Y1;  const _Float16* y1i = sy1 + (16 + t) * H_Y1;
  const _Float16* x0r = sx0 + t * H_X0;  const _Float16* x0i = sx0 + (16 + t) * H_X0;
  const _Float16* x1r = sx1 + t * H_X1;  const _Float16* x1i = sx1 + (16 + t) * H_X1;
  const h2* Y0r = dy0 + t * D_Y0;  const h2* Y0i = dy0 + (16 + t) * D_Y0;
  const h2* Y1r = dy1 + t * D_Y1;  const h2* Y1i = dy1 + (16 + t) * D_Y1;
  const h2* X0r = dx0 + t * D_X0;  const h2* X0i = dx0 + (16 + t) * D_X0;
  const h2* X1r = dx1 + t * D_X1;  const h2* X1i = dx1 + (16 + t) * D_X1;

  v8f accR[3] = {}, accI[3] = {};
  const _Float16* wrb = Wt1 + (0 * 16 + t) * 5376;
  const _Float16* wib = Wt1 + (1 * 16 + t) * 5376;
  const _Float16* wnb = Wt1 + (2 * 16 + t) * 5376;
  const int kselB = half * 8;
  const int db = half * 8;

  for (int kb = wu * 32; kb < 5376; kb += 128) {   // kb is scalar
    v16h br = load_bfrag(wrb + kb + kselB);
    v16h bi = load_bfrag(wib + kb + kselB);
    v16h bn = load_bfrag(wnb + kb + kselB);
    if (kb < 1024) {                        // yy(0,1): wide, A=y0[c], B=y1[q]
      int c = kb >> 5;
      h2 Ar = Y0r[c], Ai = Y0i[c];
#pragma unroll
      for (int q = 0; q < 3; ++q) {
        F16x16 far_, fai_;
        put8_1(far_, fai_, 0, Ar, Ai, y1r + q * 32, y1i + q * 32, db);
        put8_1(far_, fai_, 4, Ar, Ai, y1r + q * 32, y1i + q * 32, 16 + db);
        accum_c(accR[q], accI[q], far_.v, fai_.v, br, bi, bn);
      }
    } else if (kb < 2048) {                 // yy(1,0): wide, A=y1[q,c], B=y0
      int c = (kb - 1024) >> 5;
#pragma unroll
      for (int q = 0; q < 3; ++q) {
        h2 Ar = Y1r[q * 32 + c], Ai = Y1i[q * 32 + c];
        F16x16 far_, fai_;
        put8_1(far_, fai_, 0, Ar, Ai, y0r, y0i, db);
        put8_1(far_, fai_, 4, Ar, Ai, y0r, y0i, 16 + db);
        accum_c(accR[q], accI[q], far_.v, fai_.v, br, bi, bn);
      }
    } else if (kb < 3072) {                 // yy(1,1): wide 2-term
      int c = (kb - 2048) >> 5;
#pragma unroll
      for (int q = 0; q < 3; ++q) {
        const int i1 = (q == 0) ? 1 : 2, i2 = (q == 2) ? 1 : 0;
        h2 a1r = Y1r[i1 * 32 + c], a1i = Y1i[i1 * 32 + c];
        h2 a2r = Y1r[i2 * 32 + c], a2i = Y1i[i2 * 32 + c];
        F16x16 far_, fai_;
        put8_2(far_, fai_, 0, a1r, a1i, y1r + i2 * 32, y1i + i2 * 32,
                              a2r, a2i, y1r + i1 * 32, y1i + i1 * 32, db);
        put8_2(far_, fai_, 4, a1r, a1i, y1r + i2 * 32, y1i + i2 * 32,
                              a2r, a2i, y1r + i1 * 32, y1i + i1 * 32, 16 + db);
        accum_c(accR[q], accI[q], far_.v, fai_.v, br, bi, bn);
      }
    } else if (kb < 3584) {                 // yx(0,1): narrow, A=y0[c], B=x1[q]
      int c = (kb - 3072) >> 4;
      h2 A0r = Y0r[c],     A0i = Y0i[c];
      h2 A1r = Y0r[c + 1], A1i = Y0i[c + 1];
#pragma unroll
      for (int q = 0; q < 3; ++q) {
        F16x16 far_, fai_;
        put8_1(far_, fai_, 0, A0r, A0i, x1r + q * 16, x1i + q * 16, db);
        put8_1(far_, fai_, 4, A1r, A1i, x1r + q * 16, x1i + q * 16, db);
        accum_c(accR[q], accI[q], far_.v, fai_.v, br, bi, bn);
      }
    } else if (kb < 4096) {                 // yx(1,0): narrow, A=y1[q,c], B=x0
      int c = (kb - 3584) >> 4;
#pragma unroll
      for (int q = 0; q < 3; ++q) {
        F16x16 far_, fai_;
        put8_1(far_, fai_, 0, Y1r[q * 32 + c],     Y1i[q * 32 + c],     x0r, x0i, db);
        put8_1(far_, fai_, 4, Y1r[q * 32 + c + 1], Y1i[q * 32 + c + 1], x0r, x0i, db);
        accum_c(accR[q], accI[q], far_.v, fai_.v, br, bi, bn);
      }
    } else if (kb < 4608) {                 // yx(1,1): narrow 2-term
      int c = (kb - 4096) >> 4;
#pragma unroll
      for (int q = 0; q < 3; ++q) {
        const int i1 = (q == 0) ? 1 : 2, i2 = (q == 2) ? 1 : 0;
        F16x16 far_, fai_;
#pragma unroll
        for (int h = 0; h < 2; ++h) {
          int cc = c + h;
          put8_2(far_, fai_, h * 4,
                 Y1r[i1 * 32 + cc], Y1i[i1 * 32 + cc], x1r + i2 * 16, x1i + i2 * 16,
                 Y1r[i2 * 32 + cc], Y1i[i2 * 32 + cc], x1r + i1 * 16, x1i + i1 * 16,
                 db);
        }
        accum_c(accR[q], accI[q], far_.v, fai_.v, br, bi, bn);
      }
    } else if (kb < 4864) {                 // xx(0,1): narrow, A=x0[c], B=x1[q]
      int c = (kb - 4608) >> 4;
      h2 A0r = X0r[c],     A0i = X0i[c];
      h2 A1r = X0r[c + 1], A1i = X0i[c + 1];
#pragma unroll
      for (int q = 0; q < 3; ++q) {
        F16x16 far_, fai_;
        put8_1(far_, fai_, 0, A0r, A0i, x1r + q * 16, x1i + q * 16, db);
        put8_1(far_, fai_, 4, A1r, A1i, x1r + q * 16, x1i + q * 16, db);
        accum_c(accR[q], accI[q], far_.v, fai_.v, br, bi, bn);
      }
    } else if (kb < 5120) {                 // xx(1,0): narrow, A=x1[q,c], B=x0
      int c = (kb - 4864) >> 4;
#pragma unroll
      for (int q = 0; q < 3; ++q) {
        F16x16 far_, fai_;
        put8_1(far_, fai_, 0, X1r[q * 16 + c],     X1i[q * 16 + c],     x0r, x0i, db);
        put8_1(far_, fai_, 4, X1r[q * 16 + c + 1], X1i[q * 16 + c + 1], x0r, x0i, db);
        accum_c(accR[q], accI[q], far_.v, fai_.v, br, bi, bn);
      }
    } else {                                // xx(1,1): narrow 2-term
      int c = (kb - 5120) >> 4;
#pragma unroll
      for (int q = 0; q < 3; ++q) {
        const int i1 = (q == 0) ? 1 : 2, i2 = (q == 2) ? 1 : 0;
        F16x16 far_, fai_;
#pragma unroll
        for (int h = 0; h < 2; ++h) {
          int cc = c + h;
          put8_2(far_, fai_, h * 4,
                 X1r[i1 * 16 + cc], X1i[i1 * 16 + cc], x1r + i2 * 16, x1i + i2 * 16,
                 X1r[i2 * 16 + cc], X1i[i2 * 16 + cc], x1r + i1 * 16, x1i + i1 * 16,
                 db);
        }
        accum_c(accR[q], accI[q], far_.v, fai_.v, br, bi, bn);
      }
    }
  }

#pragma unroll
  for (int a = 0; a < 6; ++a) {
    v8f v8 = (a < 3) ? accR[a] : accI[a - 3];
#pragma unroll
    for (int v = 0; v < 8; ++v)
      sred[((wu * 6 + a) * 8 + v) * 32 + ln] = v8[v];
  }
  __syncthreads();
  if (wu == 0) {
#pragma unroll
    for (int a = 0; a < 6; ++a) {
      int q = a % 3, ri = a / 3;
#pragma unroll
      for (int v = 0; v < 8; ++v) {
        float s = 0.f;
#pragma unroll
        for (int ww = 0; ww < 4; ++ww) s += sred[((ww * 6 + a) * 8 + v) * 32 + ln];
        int node = nb + v + 8 * half;
        out1[((ri * NN + node) * 3 + q) * 16 + t] = s;
      }
    }
  }
}

// ---------------------------------------------------------------------------
extern "C" void kernel_launch(void* const* d_in, const int* in_sizes, int n_in,
                              void* d_out, int out_size, void* d_ws, size_t ws_size,
                              hipStream_t stream) {
  (void)in_sizes; (void)n_in; (void)out_size; (void)ws_size;
  const float* x0    = (const float*)d_in[0];
  const float* x1    = (const float*)d_in[1];
  const float* ev    = (const float*)d_in[2];
  const int*   eidx  = (const int*)d_in[3];
  const float* wii0r = (const float*)d_in[4];
  const float* wii0i = (const float*)d_in[5];
  const float* wii1r = (const float*)d_in[6];
  const float* wii1i = (const float*)d_in[7];
  const float* wmi0r = (const float*)d_in[8];
  const float* wmi0i = (const float*)d_in[9];
  const float* wmi1r = (const float*)d_in[10];
  const float* wmi1i = (const float*)d_in[11];
  const float* wmm0r = (const float*)d_in[12];
  const float* wmm0i = (const float*)d_in[13];
  const float* wmm1r = (const float*)d_in[14];
  const float* wmm1i = (const float*)d_in[15];

  char* ws = (char*)d_ws;
  float*     y0  = (float*)ws;                       // 2*N*32
  float*     y1  = y0 + 2 * NN * 32;                 // 2*N*96
  _Float16*  Wt0 = (_Float16*)(ws + 4194304);        // 3*16*3584
  _Float16*  Wt1 = Wt0 + 3 * 16 * 3584;              // 3*16*5376
  float*     out0 = (float*)d_out;
  float*     out1 = out0 + 2 * NN * 16;

  hipMemsetAsync(y0, 0, (size_t)(2 * NN * 128) * sizeof(float), stream);
  k_prep_w0<<<(3 * 16 * 3584 + 255) / 256, 256, 0, stream>>>(
      wmm0r, wmm0i, wmi0r, wmi0i, wii0r, wii0i, Wt0);
  k_prep_w1<<<(3 * 16 * 5376 + 255) / 256, 256, 0, stream>>>(
      wmm1r, wmm1i, wmi1r, wmi1i, wii1r, wii1i, Wt1);
  k_edge_mp<<<(EE * 16) / 256, 256, 0, stream>>>(x0, x1, ev, eidx, y0, y1);
  k_cg_l0<<<NN / 16, 128, 0, stream>>>(x0, x1, y0, y1, Wt0, out0);
  k_cg_l1<<<NN / 16, 128, 0, stream>>>(x0, x1, y0, y1, Wt1, out1);
}
